// CrfDecoder_87668872446449
// MI455X (gfx1250) — matrix operand order repaired and checked by reading, not compiled, not run on hardware
//
#include <hip/hip_runtime.h>
#include <math.h>

typedef __attribute__((ext_vector_type(16))) _Float16 v16h;
typedef __attribute__((ext_vector_type(8)))  _Float16 v8h;
typedef __attribute__((ext_vector_type(8)))  float    v8f;

#define B_TOT   32
#define T_LEN   1024
#define K_DIM   512
#define BPG     16          // batch rows per workgroup (one WMMA M-tile)
#define KBLK    16          // K_DIM / 32 k-steps per tile
#define A_STRIDE 520        // padded f16 row stride (520*2 = 1040B, 16B aligned)

// ---- order-preserving float <-> uint encoding (lets us use ds_max_u32) ----
__device__ __forceinline__ unsigned f2ord(float x) {
  unsigned b = __float_as_uint(x);
  return (b & 0x80000000u) ? ~b : (b | 0x80000000u);
}
__device__ __forceinline__ float ord2f(unsigned k) {
  return __uint_as_float((k & 0x80000000u) ? (k ^ 0x80000000u) : ~k);
}

// A fragment (16x32 f16 WMMA layout) from padded LDS row.
__device__ __forceinline__ v16h load_a_frag(const _Float16* base, int kb) {
  const _Float16* ap = base + (kb << 5);
  v8h lo = *(const v8h*)ap;
  v8h hi = *(const v8h*)(ap + 8);
  return __builtin_shufflevector(lo, hi,
      0, 1, 2, 3, 4, 5, 6, 7, 8, 9, 10, 11, 12, 13, 14, 15);
}

// --------------------------------------------------------------------------
// Prep: E = exp(trans) as f16 in the per-lane WMMA B-matrix (32x16) layout:
//   e_swz[((tn*KBLK + kb)*32 + lane)*16 + h]
//   column j = tn*16 + (lane&15);  K = kb*32 + (lane>=16 ? 16 : 0) + h
// --------------------------------------------------------------------------
__global__ void crf_prep_exp_trans(const float* __restrict__ trans,
                                   _Float16* __restrict__ e_swz) {
  int idx = blockIdx.x * blockDim.x + threadIdx.x;
  if (idx >= K_DIM * K_DIM) return;
  int i = idx / K_DIM;                 // source tag (reduction index)
  int j = idx % K_DIM;                 // dest tag (output column)
  float v = __expf(trans[idx]);
  int tn = j >> 4;
  int n  = j & 15;
  int kb = i >> 5;
  int ik = i & 31;
  int lane = n + ((ik & 16) ? 16 : 0);
  int h    = ik & 15;
  e_swz[(((size_t)(tn * KBLK + kb) * 32 + lane) << 4) + h] = (_Float16)v;
}

// --------------------------------------------------------------------------
// Persistent forward scan. grid = 2 workgroups (independent batch halves),
// 1024 threads = 32 wave32s; wave w owns output columns [w*16, w*16+16).
// --------------------------------------------------------------------------
__global__ __launch_bounds__(1024)
void crf_forward_persistent(const float* __restrict__ em,
                            const int*   __restrict__ lengths,
                            const float* __restrict__ head,
                            const float* __restrict__ last,
                            const _Float16* __restrict__ e_swz,
                            float* __restrict__ out) {
  __shared__ float                  alpha[BPG * K_DIM];         // 32 KB
  __shared__ __align__(16) _Float16 a16[BPG * A_STRIDE];        // ~16.3 KB
  __shared__ unsigned               rowmaxOrd[2][BPG];
  __shared__ int                    len_s[BPG];

  const int tid  = threadIdx.x;
  const int wg   = blockIdx.x;              // 0 or 1
  const int lane = tid & 31;
  const int wave = tid >> 5;                // 0..31 -> column tile
  const int b0   = wg * BPG;

  if (tid < BPG) len_s[tid] = lengths[b0 + tid];
  if (tid < 2 * BPG) ((unsigned*)rowmaxOrd)[tid] = f2ord(-INFINITY);
  __syncthreads();

  // ---- t = 0: alpha0 = head + em[:,0,:], seed rowmax[0] ----
  {
    int base = tid * 8;                     // 1024 threads * 8 = 16*512
    int r = base >> 9, k = base & 511;
    float m = -INFINITY;
    for (int e = 0; e < 8; ++e) {
      float v = head[k + e] + em[(size_t)(b0 + r) * T_LEN * K_DIM + k + e];
      alpha[base + e] = v;
      m = fmaxf(m, v);
    }
    atomicMax(&rowmaxOrd[0][r], f2ord(m));  // ds_max_u32
  }
  __syncthreads();

  const int tn   = wave;
  const int j0   = tn << 4;
  const int mrow = lane & 15;               // M row of A fragment
  const int hsel = lane >> 4;               // which K-half this lane carries
  const int jcol = j0 + (lane & 15);        // output column of C fragment

  // Per-lane loop-invariant row metadata (avoid per-step LDS reads).
  int mylen[8];
  for (int r = 0; r < 8; ++r) mylen[r] = len_s[r + (hsel << 3)];

  for (int t = 1; t < T_LEN; ++t) {
    // ---- Phase A: a16 = exp(alpha - rowmax_prev); reset rowmax_cur ----
    {
      int base = tid * 8;
      int r = base >> 9, k = base & 511;
      float m = ord2f(rowmaxOrd[(t + 1) & 1][r]);     // == (t-1)&1
      for (int e = 0; e < 8; ++e)
        a16[r * A_STRIDE + k + e] = (_Float16)__expf(alpha[base + e] - m);
      if (tid < BPG) rowmaxOrd[t & 1][tid] = f2ord(-INFINITY);
    }
    __syncthreads();

    // Prefetch next step's emission lines while the GEMM runs
    // (global_prefetch_b8; one lane per half-wave covers the 64B row chunk).
    if ((t + 1 < T_LEN) && ((lane & 15) == 0)) {
      for (int r = 0; r < 8; ++r) {
        int bl = r + (hsel << 3);
        __builtin_prefetch(
            &em[((size_t)(b0 + bl) * T_LEN + t + 1) * K_DIM + j0], 0, 3);
      }
    }

    // Opaque 32-bit element offset: defeats LICM across the t-loop (so B
    // streams from L2 each step, no scratch spill) while keeping the global
    // address-space provenance of e_swz -> SADDR global_load_b128 + imm.
    int boff = ((tn * KBLK) << 9) + (lane << 4);
    asm volatile("" : "+v"(boff));
    const _Float16* bp  = e_swz + boff;
    const _Float16* ap0 = &a16[mrow * A_STRIDE + (hsel << 4)];

    // ---- Phase B: software-pipelined 16-deep WMMA chain ----
    // B loaded 2 iterations ahead (L2 latency), A 1 ahead (LDS latency).
    v16h a_cur = load_a_frag(ap0, 0);
    v16h b_cur = *(const v16h*)(bp);
    v16h b_nxt = *(const v16h*)(bp + (1 << 9));
    v8f acc = {};
#pragma unroll
    for (int kb = 0; kb < KBLK; ++kb) {
      v16h a_nxt = a_cur;
      v16h b_fut = b_cur;
      if (kb + 1 < KBLK) a_nxt = load_a_frag(ap0, kb + 1);
      if (kb + 2 < KBLK) b_fut = *(const v16h*)(bp + ((kb + 2) << 9));
      acc = __builtin_amdgcn_wmma_f32_16x16x32_f16(
          false, a_cur, false, b_cur, (short)0, acc, false, false);
      a_cur = a_nxt;
      b_cur = b_nxt;
      b_nxt = b_fut;
    }

    // ---- Epilogue: alpha_new = m + log(acc) + em, ragged freeze ----
    float nv_arr[8];
    for (int r = 0; r < 8; ++r) {
      int bl = r + (hsel << 3);             // C layout: M = r (+8 for hi lanes)
      float old = alpha[bl * K_DIM + jcol];
      float nv;
      if (t < mylen[r]) {
        float m = ord2f(rowmaxOrd[(t + 1) & 1][bl]);
        nv = m + __logf(acc[r]) +
             em[((size_t)(b0 + bl) * T_LEN + t) * K_DIM + jcol];
      } else {
        nv = old;                           // frozen past sequence end
      }
      alpha[bl * K_DIM + jcol] = nv;
      nv_arr[r] = nv;
    }
    // Row-max: shuffle-reduce across the 16-lane half that owns these rows,
    // then a single lane per half issues the ds_max_u32 (16x fewer atomics).
    for (int r = 0; r < 8; ++r) {
      float m = nv_arr[r];
      m = fmaxf(m, __shfl_xor(m, 1, 32));
      m = fmaxf(m, __shfl_xor(m, 2, 32));
      m = fmaxf(m, __shfl_xor(m, 4, 32));
      m = fmaxf(m, __shfl_xor(m, 8, 32));
      nv_arr[r] = m;
    }
    if ((lane & 15) == 0) {
      for (int r = 0; r < 8; ++r)
        atomicMax(&rowmaxOrd[t & 1][r + (hsel << 3)], f2ord(nv_arr[r]));
    }
    __syncthreads();
  }

  // ---- Z[b] = logsumexp_j(alpha[b][j] + last[j]) : one wave per row ----
  if (wave < BPG) {
    int r = wave;
    float vals[16];
    float vmax = -INFINITY;
    for (int q = 0; q < 16; ++q) {
      int k = (q << 5) + lane;
      vals[q] = alpha[r * K_DIM + k] + last[k];
      vmax = fmaxf(vmax, vals[q]);
    }
    for (int off = 16; off; off >>= 1)
      vmax = fmaxf(vmax, __shfl_xor(vmax, off, 32));
    float s = 0.f;
    for (int q = 0; q < 16; ++q) s += __expf(vals[q] - vmax);
    for (int off = 16; off; off >>= 1)
      s += __shfl_xor(s, off, 32);
    if (lane == 0) out[b0 + r] = vmax + __logf(s);
  }
}

// --------------------------------------------------------------------------
extern "C" void kernel_launch(void* const* d_in, const int* in_sizes, int n_in,
                              void* d_out, int out_size, void* d_ws, size_t ws_size,
                              hipStream_t stream) {
  const float* em    = (const float*)d_in[0];   // [B,T,K] f32
  const int*   len   = (const int*)  d_in[1];   // [B] i32
  const float* trans = (const float*)d_in[2];   // [K,K] f32
  const float* head  = (const float*)d_in[3];   // [K] f32
  const float* lastT = (const float*)d_in[4];   // [K] f32
  float* out = (float*)d_out;                   // [B] f32

  _Float16* e_swz = (_Float16*)d_ws;            // 512 KB swizzled exp(trans)

  crf_prep_exp_trans<<<(K_DIM * K_DIM + 255) / 256, 256, 0, stream>>>(trans, e_swz);
  crf_forward_persistent<<<2, 1024, 0, stream>>>(em, len, head, lastT, e_swz, out);
}
